// StaticSparseAttention_17360257810615
// MI455X (gfx1250) — matrix operand order repaired and checked by reading, not compiled, not run on hardware
//
#include <hip/hip_runtime.h>
#include <cstdint>

#define DEVFN __device__ __forceinline__

typedef __bf16 bf16;
typedef __attribute__((ext_vector_type(16))) __bf16 v16bf;
typedef __attribute__((ext_vector_type(8)))  __bf16 bf16x8;
typedef __attribute__((ext_vector_type(8)))  float   v8f;

// ---------------- problem constants ----------------
constexpr int S_TOT  = 8196;   // NUM_CLS + PATCH_LEN
constexpr int EMBED  = 512;
constexpr int H      = 8;
constexpr int HD     = 64;
constexpr int NCLS   = 4;
constexpr int PATCH  = 8192;
constexpr int NQKV   = 1536;
constexpr int MPAD   = 8256;   // 129 * 64 row padding for GEMM M
constexpr int KEYLEN = 8352;   // padded key axis: idx = (s<4)? s : s+28  (patch p at 32+p)
constexpr float SCALE  = 0.125f;   // 1/sqrt(64)
constexpr float NEGINF = -1e30f;

// ---------------- WMMA helpers ----------------
DEVFN v8f wmma_bf16(v16bf a, v16bf b, v8f c) {
#if defined(__HIP_DEVICE_COMPILE__)
  return __builtin_amdgcn_wmma_f32_16x16x32_bf16(false, a, false, b, (short)0, c,
                                                 false, false);
#else
  return c;
#endif
}

DEVFN v8f zero8() {
  v8f z;
#pragma unroll
  for (int i = 0; i < 8; ++i) z[i] = 0.0f;
  return z;
}

// A fragment (16x32 bf16, row-major MxK source). lane L: row L&15;
// lanes 0-15 -> K 0..15, lanes 16-31 -> K 16..31.
DEVFN v16bf load_fragA(const bf16* __restrict__ base, int ld, int k0, int lane) {
  int row = lane & 15;
  const bf16* p = base + (long)row * ld + k0 + ((lane >> 4) << 4);
  bf16x8 a0 = *(const bf16x8*)(p);
  bf16x8 a1 = *(const bf16x8*)(p + 8);
  v16bf r;
#pragma unroll
  for (int i = 0; i < 8; ++i) { r[i] = a0[i]; r[8 + i] = a1[i]; }
  return r;
}

// B fragment (32x16 bf16) from NxK row-major source (row n = column n of B).
// lanes 0-15: K=0..15, lanes 16-31: K=16..31 -> single contiguous 32B run.
DEVFN v16bf load_fragB(const bf16* __restrict__ base, int ld, int k0, int lane) {
  int row = lane & 15;
  const bf16* p = base + (long)row * ld + k0 + ((lane >> 4) << 4);
  return *(const v16bf*)(p);
}

DEVFN void split_store(bf16* hi, bf16* lo, float v) {
  bf16 h = (bf16)v;
  *hi = h;
  *lo = (bf16)(v - (float)h);
}

// ---------------- K1: fp32 -> bf16 hi/lo split ----------------
__global__ void __launch_bounds__(256)
split_kernel(const float* __restrict__ src, bf16* __restrict__ hi,
             bf16* __restrict__ lo, long n) {
  long i = (long)blockIdx.x * 256 + threadIdx.x;
  if (i < n) {
    float v = src[i];
    bf16 h = (bf16)v;
    hi[i] = h;
    lo[i] = (bf16)(v - (float)h);
  }
}

// ---------------- shared GEMM core: one wave -> 16(M) x 64(N), K=512 ----------------
DEVFN void gemm_core(const bf16* __restrict__ Ah, const bf16* __restrict__ Al,
                     const bf16* __restrict__ Bh, const bf16* __restrict__ Bl,
                     int lane, v8f acc[4]) {
  for (int k0 = 0; k0 < 512; k0 += 32) {
#if defined(__HIP_DEVICE_COMPILE__)
    if (k0 + 32 < 512) __builtin_prefetch(Ah + (long)(lane & 15) * 512 + k0 + 32, 0, 1);
#endif
    v16bf ah = load_fragA(Ah, 512, k0, lane);
    v16bf al = load_fragA(Al, 512, k0, lane);
#pragma unroll
    for (int s = 0; s < 4; ++s) {
      v16bf bh = load_fragB(Bh + (long)s * 16 * 512, 512, k0, lane);
      v16bf bl = load_fragB(Bl + (long)s * 16 * 512, 512, k0, lane);
      acc[s] = wmma_bf16(ah, bh, acc[s]);
      acc[s] = wmma_bf16(ah, bl, acc[s]);
      acc[s] = wmma_bf16(al, bh, acc[s]);
    }
  }
}

// ---------------- K2: QKV GEMM (x @ w_qkv.T + b) with routed epilogue ----------------
// q,k -> fp32 [h][MPAD][64]; v -> bf16 hi/lo transposed vT[h][64][KEYLEN]
__global__ void __launch_bounds__(128)
gemm_qkv_kernel(const bf16* __restrict__ xhi, const bf16* __restrict__ xlo,
                const bf16* __restrict__ whi, const bf16* __restrict__ wlo,
                const float* __restrict__ bias,
                float* __restrict__ qf, float* __restrict__ kf,
                bf16* __restrict__ vthi, bf16* __restrict__ vtlo) {
  int lane = threadIdx.x & 31;
  int wv   = threadIdx.x >> 5;
  int m0   = blockIdx.x * 64 + wv * 16;
  int n0   = blockIdx.y * 64;

  v8f acc[4];
#pragma unroll
  for (int s = 0; s < 4; ++s) acc[s] = zero8();

  gemm_core(xhi + (long)m0 * 512, xlo + (long)m0 * 512,
            whi + (long)n0 * 512, wlo + (long)n0 * 512, lane, acc);

  int half = lane >> 4, nl = lane & 15;
#pragma unroll
  for (int s = 0; s < 4; ++s) {
    int ng = n0 + s * 16 + nl;
    float bv = bias[ng];
    int t  = ng >> 9;
    int hh = (ng & 511) >> 6;
    int d  = ng & 63;
#pragma unroll
    for (int r = 0; r < 8; ++r) {
      int m = m0 + half * 8 + r;  // global token row (pads write junk in-bounds, never read)
      float v = acc[s][r] + bv;
      if (t == 0) {
        qf[((long)hh * MPAD + m) * 64 + d] = v;
      } else if (t == 1) {
        kf[((long)hh * MPAD + m) * 64 + d] = v;
      } else {
        long idx = (m < NCLS) ? m : (m + 28);
        if (idx < KEYLEN) {
          long o = ((long)hh * 64 + d) * KEYLEN + idx;
          split_store(vthi + o, vtlo + o, v);
        }
      }
    }
  }
}

// ---------------- K3: RoPE + pack q/k into bf16 hi/lo attention layouts ----------------
__global__ void __launch_bounds__(256)
rope_pack_kernel(const float* __restrict__ qf, const float* __restrict__ kf,
                 const float* __restrict__ coords,
                 bf16* __restrict__ qphi, bf16* __restrict__ qplo,
                 bf16* __restrict__ kbhi, bf16* __restrict__ kblo) {
  int idx = blockIdx.x * 256 + threadIdx.x;  // (h, s, pair) flattened: 8*8196*32
  if (idx >= H * S_TOT * 32) return;
  int i  = idx & 31;       // dim pair 0..31
  int hs = idx >> 5;
  int s  = hs % S_TOT;
  int h  = hs / S_TOT;

  long qo_in = ((long)h * MPAD + s) * 64 + 2 * i;
  float q0 = qf[qo_in], q1 = qf[qo_in + 1];
  float k0 = kf[qo_in], k1 = kf[qo_in + 1];

  if (s >= NCLS) {
    int p = s - NCLS;
    float c = (i < 16) ? coords[2 * p] : coords[2 * p + 1];
    int j = (i < 16) ? i : (i - 16);
    // inv_freq[j] = THETA^{-(2j)/32} = 2^{-(j/16)*log2(1e4)}
    float ang = (c * 1e-5f) * exp2f(-(float)j * (13.28771237954945f / 16.0f));
    float sn, cs;
    __sincosf(ang, &sn, &cs);
    float nq0 = q0 * cs - q1 * sn, nq1 = q1 * cs + q0 * sn;
    float nk0 = k0 * cs - k1 * sn, nk1 = k1 * cs + k0 * sn;
    long qo = ((long)h * PATCH + p) * 64 + 2 * i;
    split_store(qphi + qo,     qplo + qo,     nq0);
    split_store(qphi + qo + 1, qplo + qo + 1, nq1);
    long ko = ((long)h * KEYLEN + (32 + p)) * 64 + 2 * i;
    split_store(kbhi + ko,     kblo + ko,     nk0);
    split_store(kbhi + ko + 1, kblo + ko + 1, nk1);
  } else {
    long ko = ((long)h * KEYLEN + s) * 64 + 2 * i;  // cls keys: no RoPE
    split_store(kbhi + ko,     kblo + ko,     k0);
    split_store(kbhi + ko + 1, kblo + ko + 1, k1);
  }
}

// ---------------- K4a: cls attention partials (online softmax over key chunks) ----------------
__global__ void __launch_bounds__(128)
cls_partial_kernel(const float* __restrict__ qf,
                   const bf16* __restrict__ kbhi, const bf16* __restrict__ kblo,
                   const bf16* __restrict__ vthi, const bf16* __restrict__ vtlo,
                   float* __restrict__ pm, float* __restrict__ ps,
                   float* __restrict__ pa) {
  int h = blockIdx.x, chunk = blockIdx.y;
  int c = threadIdx.x >> 5, lane = threadIdx.x & 31;
  int s0 = chunk * 257;
  int s1 = s0 + 257; if (s1 > S_TOT) s1 = S_TOT;

  const float* q = qf + ((long)h * MPAD + c) * 64;
  float q0 = q[lane], q1 = q[lane + 32];
  const bf16* kh  = kbhi + (long)h * KEYLEN * 64;
  const bf16* kl  = kblo + (long)h * KEYLEN * 64;
  const bf16* vh0 = vthi + ((long)h * 64 + lane) * KEYLEN;
  const bf16* vl0 = vtlo + ((long)h * 64 + lane) * KEYLEN;
  const bf16* vh1 = vthi + ((long)h * 64 + lane + 32) * KEYLEN;
  const bf16* vl1 = vtlo + ((long)h * 64 + lane + 32) * KEYLEN;

  float m = NEGINF, sum = 0.f, a0 = 0.f, a1 = 0.f;
  for (int s = s0; s < s1; ++s) {
    long idx = (s < NCLS) ? s : (s + 28);
    const bf16* kr  = kh + idx * 64;
    const bf16* krl = kl + idx * 64;
    float sc = q0 * ((float)kr[lane]      + (float)krl[lane]) +
               q1 * ((float)kr[lane + 32] + (float)krl[lane + 32]);
#pragma unroll
    for (int msk = 16; msk >= 1; msk >>= 1) sc += __shfl_xor(sc, msk);
    sc *= SCALE;
    float mn   = fmaxf(m, sc);
    float corr = __expf(m - mn);
    float p    = __expf(sc - mn);
    sum = sum * corr + p;
    a0  = a0  * corr + p * ((float)vh0[idx] + (float)vl0[idx]);
    a1  = a1  * corr + p * ((float)vh1[idx] + (float)vl1[idx]);
    m = mn;
  }
  long base = ((long)h * 4 + c) * 32 + chunk;
  if (lane == 0) { pm[base] = m; ps[base] = sum; }
  pa[base * 64 + lane]      = a0;
  pa[base * 64 + lane + 32] = a1;
}

// ---------------- K4b: combine cls partials -> ao rows 0..3 ----------------
__global__ void __launch_bounds__(128)
cls_combine_kernel(const float* __restrict__ pm, const float* __restrict__ ps,
                   const float* __restrict__ pa, float* __restrict__ ao) {
  int h = blockIdx.x;
  int c = threadIdx.x >> 5, lane = threadIdx.x & 31;
  long base = ((long)h * 4 + c) * 32;
  float M = NEGINF;
  for (int k = 0; k < 32; ++k) M = fmaxf(M, pm[base + k]);
  float S = 0.f, A0 = 0.f, A1 = 0.f;
  for (int k = 0; k < 32; ++k) {
    float w = __expf(pm[base + k] - M);
    S  += ps[base + k] * w;
    A0 += pa[(base + k) * 64 + lane]      * w;
    A1 += pa[(base + k) * 64 + lane + 32] * w;
  }
  float inv = 1.0f / S;
  ao[(long)c * EMBED + h * 64 + lane]      = A0 * inv;
  ao[(long)c * EMBED + h * 64 + lane + 32] = A1 * inv;
}

// ---------------- K5: banded patch attention, fully WMMA ----------------
// grid (128 query-blocks, 8 heads); block = 4 waves, wave -> 16 queries.
__global__ void __launch_bounds__(128)
patch_attn_kernel(const bf16* __restrict__ qphi, const bf16* __restrict__ qplo,
                  const bf16* __restrict__ kbhi, const bf16* __restrict__ kblo,
                  const bf16* __restrict__ vthi, const bf16* __restrict__ vtlo,
                  float* __restrict__ ao) {
  __shared__ float plds_all[4][16][160];  // per-wave P tile (144 keys + 16 zero pad)
  int lane = threadIdx.x & 31;
  int wv   = threadIdx.x >> 5;
  int h    = blockIdx.y;
  int l0   = blockIdx.x * 64;   // query block base (patch index)
  int lw   = l0 + wv * 16;
  int half = lane >> 4, nl = lane & 15;

  const bf16* qh = qphi + ((long)h * PATCH + lw) * 64;
  const bf16* ql = qplo + ((long)h * PATCH + lw) * 64;
  const bf16* kh = kbhi + (long)h * KEYLEN * 64;
  const bf16* kl = kblo + (long)h * KEYLEN * 64;

  // ---- scores: 16 queries x 144 keys (tile 0 = cls slots, tiles 1..8 = window) ----
  v8f sc[9];
#pragma unroll
  for (int jt = 0; jt < 9; ++jt) sc[jt] = zero8();

#pragma unroll
  for (int k0 = 0; k0 < 64; k0 += 32) {
    v16bf ah = load_fragA(qh, 64, k0, lane);
    v16bf al = load_fragA(ql, 64, k0, lane);
#pragma unroll
    for (int jt = 0; jt < 9; ++jt) {
      long rowbase = (jt == 0) ? 0 : (long)(l0 + (jt - 1) * 16);
      v16bf bh = load_fragB(kh + rowbase * 64, 64, k0, lane);
      v16bf bl = load_fragB(kl + rowbase * 64, 64, k0, lane);
      sc[jt] = wmma_bf16(ah, bh, sc[jt]);
      sc[jt] = wmma_bf16(ah, bl, sc[jt]);
      sc[jt] = wmma_bf16(al, bh, sc[jt]);
    }
  }

  // ---- scale + band/bounds mask ----
#pragma unroll
  for (int jt = 0; jt < 9; ++jt) {
#pragma unroll
    for (int r = 0; r < 8; ++r) {
      int m  = half * 8 + r;
      int qi = wv * 16 + m;           // query index within 64-block
      bool valid;
      if (jt == 0) {
        valid = (nl < NCLS);
      } else {
        int j = (jt - 1) * 16 + nl;   // window offset 0..127
        int p = l0 - 32 + j;
        valid = (j >= qi) && (j <= qi + 64) && (p >= 0) && (p < PATCH);
      }
      float v = sc[jt][r] * SCALE;
      sc[jt][r] = valid ? v : NEGINF;
    }
  }

  // ---- row softmax (rows live across 16 lanes of each half-wave) ----
  float rinv[8];
#pragma unroll
  for (int r = 0; r < 8; ++r) {
    float mx = NEGINF;
#pragma unroll
    for (int jt = 0; jt < 9; ++jt) mx = fmaxf(mx, sc[jt][r]);
#pragma unroll
    for (int msk = 8; msk >= 1; msk >>= 1) mx = fmaxf(mx, __shfl_xor(mx, msk));
    float s = 0.f;
#pragma unroll
    for (int jt = 0; jt < 9; ++jt) {
      float p = __expf(sc[jt][r] - mx);
      sc[jt][r] = p;
      s += p;
    }
#pragma unroll
    for (int msk = 8; msk >= 1; msk >>= 1) s += __shfl_xor(s, msk);
    rinv[r] = 1.0f / s;
  }

  // ---- transpose P into A-fragment layout via wave-private LDS ----
  float (*plds)[160] = plds_all[wv];
#pragma unroll
  for (int r = 0; r < 8; ++r) {
    int m = half * 8 + r;
#pragma unroll
    for (int jt = 0; jt < 9; ++jt) plds[m][jt * 16 + nl] = sc[jt][r];
    plds[m][144 + nl] = 0.0f;       // zero pad keys 144..159
  }

  // ---- P @ V : 5 K-chunks x 4 N-tiles, P split hi/lo ----
  v8f outv[4];
#pragma unroll
  for (int s = 0; s < 4; ++s) outv[s] = zero8();
  const bf16* vt  = vthi + (long)h * 64 * KEYLEN;
  const bf16* vtl = vtlo + (long)h * 64 * KEYLEN;

#pragma unroll
  for (int c = 0; c < 5; ++c) {
    v16bf ph, pl;
    {
      const float* pr = &plds[nl][0];
      int kk0 = c * 32 + half * 16;
      int kk1 = kk0 + 8;
#pragma unroll
      for (int i = 0; i < 8; ++i) {
        float v0 = pr[kk0 + i];
        bf16 h0 = (bf16)v0; ph[i] = h0;     pl[i] = (bf16)(v0 - (float)h0);
        float v1 = pr[kk1 + i];
        bf16 h1 = (bf16)v1; ph[8 + i] = h1; pl[8 + i] = (bf16)(v1 - (float)h1);
      }
    }
    int kk = c * 32 + (half << 4);                       // 16-key run start
    long sidx = (kk < 16) ? (long)kk : (long)(l0 + kk - 16);
#pragma unroll
    for (int s = 0; s < 4; ++s) {
      int drow = s * 16 + nl;
      v16bf bh = *(const v16bf*)(vt  + (long)drow * KEYLEN + sidx);
      v16bf bl = *(const v16bf*)(vtl + (long)drow * KEYLEN + sidx);
      outv[s] = wmma_bf16(ph, bh, outv[s]);
      outv[s] = wmma_bf16(ph, bl, outv[s]);
      outv[s] = wmma_bf16(pl, bh, outv[s]);
    }
  }

  // ---- normalize + store to ao[4+l][h*64+d] ----
#pragma unroll
  for (int s = 0; s < 4; ++s) {
#pragma unroll
    for (int r = 0; r < 8; ++r) {
      int m = half * 8 + r;
      int l = lw + m;
      ao[(long)(NCLS + l) * EMBED + h * 64 + s * 16 + nl] = outv[s][r] * rinv[r];
    }
  }
}

// ---------------- K7: output projection GEMM -> d_out ----------------
__global__ void __launch_bounds__(128)
gemm_out_kernel(const bf16* __restrict__ ahi, const bf16* __restrict__ alo,
                const bf16* __restrict__ whi, const bf16* __restrict__ wlo,
                const float* __restrict__ bias, float* __restrict__ out) {
  int lane = threadIdx.x & 31;
  int wv   = threadIdx.x >> 5;
  int m0   = blockIdx.x * 64 + wv * 16;
  int n0   = blockIdx.y * 64;

  v8f acc[4];
#pragma unroll
  for (int s = 0; s < 4; ++s) acc[s] = zero8();

  gemm_core(ahi + (long)m0 * 512, alo + (long)m0 * 512,
            whi + (long)n0 * 512, wlo + (long)n0 * 512, lane, acc);

  int half = lane >> 4, nl = lane & 15;
#pragma unroll
  for (int s = 0; s < 4; ++s) {
    int ng = n0 + s * 16 + nl;
    float bv = bias[ng];
#pragma unroll
    for (int r = 0; r < 8; ++r) {
      int m = m0 + half * 8 + r;
      if (m < S_TOT) out[(long)m * EMBED + ng] = acc[s][r] + bv;
    }
  }
}

// ---------------- host launch ----------------
extern "C" void kernel_launch(void* const* d_in, const int* in_sizes, int n_in,
                              void* d_out, int out_size, void* d_ws, size_t ws_size,
                              hipStream_t stream) {
  const float* x      = (const float*)d_in[0];
  const float* coords = (const float*)d_in[1];
  const float* w_qkv  = (const float*)d_in[2];
  const float* b_qkv  = (const float*)d_in[3];
  const float* w_out  = (const float*)d_in[4];
  const float* b_out  = (const float*)d_in[5];
  float* out = (float*)d_out;

  size_t off = 0;
  auto carve = [&](size_t bytes) -> void* {
    void* p = (char*)d_ws + off;
    off += (bytes + 255) & ~(size_t)255;
    return p;
  };

  bf16* xhi  = (bf16*)carve((size_t)MPAD * 512 * 2);
  bf16* xlo  = (bf16*)carve((size_t)MPAD * 512 * 2);
  bf16* wqhi = (bf16*)carve((size_t)NQKV * 512 * 2);
  bf16* wqlo = (bf16*)carve((size_t)NQKV * 512 * 2);
  bf16* wohi = (bf16*)carve((size_t)512 * 512 * 2);
  bf16* wolo = (bf16*)carve((size_t)512 * 512 * 2);
  float* qf  = (float*)carve((size_t)H * MPAD * 64 * 4);
  float* kf  = (float*)carve((size_t)H * MPAD * 64 * 4);
  bf16* qphi = (bf16*)carve((size_t)H * PATCH * 64 * 2);
  bf16* qplo = (bf16*)carve((size_t)H * PATCH * 64 * 2);
  bf16* kbhi = (bf16*)carve((size_t)H * KEYLEN * 64 * 2);
  bf16* kblo = (bf16*)carve((size_t)H * KEYLEN * 64 * 2);
  bf16* vthi = (bf16*)carve((size_t)H * 64 * KEYLEN * 2);
  bf16* vtlo = (bf16*)carve((size_t)H * 64 * KEYLEN * 2);
  float* ao  = (float*)carve((size_t)MPAD * 512 * 4);
  bf16* aohi = (bf16*)carve((size_t)MPAD * 512 * 2);
  bf16* aolo = (bf16*)carve((size_t)MPAD * 512 * 2);
  float* pm  = (float*)carve((size_t)H * 4 * 32 * 4);
  float* ps  = (float*)carve((size_t)H * 4 * 32 * 4);
  float* pa  = (float*)carve((size_t)H * 4 * 32 * 64 * 4);
  (void)ws_size; (void)in_sizes; (void)n_in; (void)out_size;

  // 1) split inputs to bf16 hi/lo
  {
    long n = (long)S_TOT * 512;
    split_kernel<<<dim3((n + 255) / 256), dim3(256), 0, stream>>>(x, xhi, xlo, n);
  }
  {
    long n = (long)NQKV * 512;
    split_kernel<<<dim3((n + 255) / 256), dim3(256), 0, stream>>>(w_qkv, wqhi, wqlo, n);
  }
  {
    long n = (long)512 * 512;
    split_kernel<<<dim3((n + 255) / 256), dim3(256), 0, stream>>>(w_out, wohi, wolo, n);
  }

  // 2) QKV projection
  gemm_qkv_kernel<<<dim3(MPAD / 64, NQKV / 64), dim3(128), 0, stream>>>(
      xhi, xlo, wqhi, wqlo, b_qkv, qf, kf, vthi, vtlo);

  // 3) RoPE + pack
  {
    int total = H * S_TOT * 32;
    rope_pack_kernel<<<dim3((total + 255) / 256), dim3(256), 0, stream>>>(
        qf, kf, coords, qphi, qplo, kbhi, kblo);
  }

  // 4) cls attention (partials + combine)
  cls_partial_kernel<<<dim3(H, 32), dim3(128), 0, stream>>>(
      qf, kbhi, kblo, vthi, vtlo, pm, ps, pa);
  cls_combine_kernel<<<dim3(H), dim3(128), 0, stream>>>(pm, ps, pa, ao);

  // 5) banded patch attention
  patch_attn_kernel<<<dim3(PATCH / 64, H), dim3(128), 0, stream>>>(
      qphi, qplo, kbhi, kblo, vthi, vtlo, ao);

  // 6) split attention output, 7) output projection
  {
    long n = (long)S_TOT * 512;
    split_kernel<<<dim3((n + 255) / 256), dim3(256), 0, stream>>>(ao, aohi, aolo, n);
  }
  gemm_out_kernel<<<dim3(MPAD / 64, EMBED / 64), dim3(128), 0, stream>>>(
      aohi, aolo, wohi, wolo, b_out, out);
}
